// TransformerClassifier_38680475467966
// MI455X (gfx1250) — matrix operand order repaired and checked by reading, not compiled, not run on hardware
//
#include <hip/hip_runtime.h>
#include <hip/hip_bf16.h>

typedef __attribute__((ext_vector_type(16))) _Float16 v16h;
typedef __attribute__((ext_vector_type(8)))  _Float16 v8h;
typedef __attribute__((ext_vector_type(8)))  float    v8f;

#define BATCH 64
#define TN    768
#define DIM   768
#define ATTR  51
#define KCH   24          // 768 / 32
#define NT    48          // 768 / 16 column tiles
#define CPW   6           // column tiles per wave (8 waves * 6 = 48)
#define THRESH 0.75f
#define EPS   1e-5f

// ---------------------------------------------------------------------------
// K1: per-token L2 normalize -> f16, store norm as score. One wave per token.
// ---------------------------------------------------------------------------
__global__ void k1_normalize(const float* __restrict__ x,
                             _Float16* __restrict__ xn,
                             float* __restrict__ scores) {
    int token = blockIdx.x * 8 + (threadIdx.x >> 5);
    int lane  = threadIdx.x & 31;
    const float* xr = x + (size_t)token * DIM;
    float v[24];
    float ss = 0.f;
#pragma unroll
    for (int i = 0; i < 24; ++i) { v[i] = xr[lane + 32 * i]; ss += v[i] * v[i]; }
#pragma unroll
    for (int off = 16; off > 0; off >>= 1) ss += __shfl_xor(ss, off, 32);
    float norm = sqrtf(ss);
    float inv  = 1.0f / norm;
    _Float16* o = xn + (size_t)token * DIM;
#pragma unroll
    for (int i = 0; i < 24; ++i) o[lane + 32 * i] = (_Float16)(v[i] * inv);
    if (lane == 0) scores[token] = norm;
}

// ---------------------------------------------------------------------------
// K2: similarity Gram tiles via v_wmma_f32_16x16x32_f16, threshold -> keep.
// Block: 256 threads (8 waves), one 16-row tile; A staged in LDS and shared
// by all waves. Each wave keeps 6 column-tile accumulators live, so each A
// fragment (2 ds_load_b128 per K-chunk) feeds 6 WMMAs, and the 6 independent
// B-load->WMMA chains overlap memory with XDL latency. kc loop kept rolled
// to avoid the 24-fragment register blowup/spill seen previously.
// ---------------------------------------------------------------------------
__global__ void k2_sim_keep(const _Float16* __restrict__ xn,
                            int* __restrict__ keep) {
    __shared__ _Float16 As[16 * DIM];   // 24 KB
    __shared__ int flagShared;

    const int b   = blockIdx.x / NT;
    const int rt  = blockIdx.x % NT;
    const int tid = threadIdx.x;
    const int wid = tid >> 5;
    const int lane = tid & 31;
    const int rowbase = rt * 16;

    if (tid == 0) flagShared = 0;

    // cooperative load of A tile: 16 rows x 768 halves = 1536 uint4
    const _Float16* gA = xn + ((size_t)b * TN + rowbase) * DIM;
    const uint4* gA4 = (const uint4*)gA;
    uint4* As4 = (uint4*)As;
#pragma unroll
    for (int i = 0; i < 6; ++i) As4[tid + 256 * i] = gA4[tid + 256 * i];
    __syncthreads();

    const _Float16* xnB = xn + (size_t)b * TN * DIM;
    const int mrow  = lane & 15;
    const int aoff  = (lane < 16) ? 0 : 8;    // K sub-block per ISA A layout
    const int boff  = (lane < 16) ? 0 : 8;    // K sub-block per ISA B layout
    // wave wid handles column tiles ct = wid + 8*j, j = 0..5
    // column for fragment j: (wid + 8*j)*16 + mrow = wid*16 + mrow + j*128
    const _Float16* bbase = xnB + (size_t)(wid * 16 + mrow) * DIM + boff;

    v8f acc[CPW];
#pragma unroll
    for (int j = 0; j < CPW; ++j) acc[j] = (v8f){};

#pragma unroll 1
    for (int kc = 0; kc < KCH; ++kc) {
        const int kbase = kc * 32;
        // A fragment: two 16B LDS reads (lanes<16: K+0..7/+16..23, lanes>=16: +8..15/+24..31)
        v8h a0 = *(const v8h*)(As + mrow * DIM + kbase + aoff);
        v8h a1 = *(const v8h*)(As + mrow * DIM + kbase + aoff + 16);
        v16h av = __builtin_shufflevector(a0, a1,
            0,1,2,3,4,5,6,7,8,9,10,11,12,13,14,15);
#pragma unroll
        for (int j = 0; j < CPW; ++j) {
            const _Float16* bp = bbase + (size_t)j * 128 * DIM + kbase;
            v8h b0 = *(const v8h*)(bp);
            v8h b1 = *(const v8h*)(bp + 16);
            v16h bv = __builtin_shufflevector(b0, b1,
                0,1,2,3,4,5,6,7,8,9,10,11,12,13,14,15);
            acc[j] = __builtin_amdgcn_wmma_f32_16x16x32_f16(
                false, av, false, bv, (short)0, acc[j], false, false);
        }
    }

    // threshold + diagonal exclusion; ballot-reduce columns per row
    unsigned waveAny = 0;   // bitmask over the 16 rows (uniform per wave)
#pragma unroll
    for (int j = 0; j < CPW; ++j) {
        const int ncol = (wid + 8 * j) * 16 + mrow;
#pragma unroll
        for (int v = 0; v < 8; ++v) {
            int rglob = rowbase + v + ((lane >= 16) ? 8 : 0);
            bool pred = (acc[j][v] > THRESH) && (ncol != rglob);
            unsigned m2 = __builtin_amdgcn_ballot_w32(pred);
            if (m2 & 0xFFFFu) waveAny |= (1u << v);        // row v (lanes 0-15)
            if (m2 >> 16)     waveAny |= (1u << (v + 8));  // row v+8 (lanes 16-31)
        }
    }
    if (lane == 0) atomicOr(&flagShared, (int)waveAny);
    __syncthreads();
    if (tid < 16)
        keep[b * TN + rowbase + tid] = ((flagShared >> tid) & 1) ^ 1;
}

// ---------------------------------------------------------------------------
// K3: per-sample count, exact top-128 fallback (rank matches lax.top_k
// tie-break by lower index), stable compaction index list, global max length.
// One 256-thread block per sample.
// ---------------------------------------------------------------------------
__global__ void k3_compact(const float* __restrict__ scores,
                           const int* __restrict__ keepIn,
                           int* __restrict__ idx,
                           int* __restrict__ counts,
                           int* __restrict__ Lmax) {
    __shared__ float s[TN];
    __shared__ int   k[TN];
    __shared__ int   cnt;
    const int b = blockIdx.x, tid = threadIdx.x;
    if (tid == 0) cnt = 0;
    __syncthreads();
    int local = 0;
    for (int i = tid; i < TN; i += 256) {
        s[i] = scores[b * TN + i];
        int kv = keepIn[b * TN + i];
        k[i] = kv; local += kv;
    }
    atomicAdd(&cnt, local);
    __syncthreads();
    if (cnt < 128) {                       // uniform branch
        for (int i = tid; i < TN; i += 256) {
            float si = s[i]; int rank = 0;
            for (int j = 0; j < TN; ++j) {
                float sj = s[j];
                rank += (sj > si) || (sj == si && j < i);
            }
            k[i] = (rank < 128) ? 1 : 0;
        }
    }
    __syncthreads();
    if (tid == 0) {
        int pos = 0;
        for (int i = 0; i < TN; ++i)
            if (k[i]) idx[b * TN + pos++] = i;
        counts[b] = pos;
        atomicMax(Lmax, pos);
    }
}

// ---------------------------------------------------------------------------
// K4: adaptive-pool bin gather + LayerNorm + attribute head dot.
// One wave per (b, a); only the 51 pooled positions that feed the heads.
// ---------------------------------------------------------------------------
__global__ void k4_head(const float* __restrict__ x,
                        const int* __restrict__ idx,
                        const int* __restrict__ counts,
                        const int* __restrict__ Lmax,
                        const float* __restrict__ ln_g,
                        const float* __restrict__ ln_b,
                        const float* __restrict__ w,
                        const float* __restrict__ bb,
                        float* __restrict__ logits) {
    const int b = blockIdx.x / ATTR;
    const int a = blockIdx.x % ATTR;
    const int lane = threadIdx.x;
    const int L = *Lmax;
    const int count = counts[b];
    const int start = (a * L) >> 7;                 // (a*L)/128
    const int end   = ((a + 1) * L + 127) >> 7;     // ceil
    const float invw = 1.0f / (float)(end - start);
    const int pend = min(end, count);

    float acc[24];
#pragma unroll
    for (int i = 0; i < 24; ++i) acc[i] = 0.f;
    for (int p = start; p < pend; ++p) {
        int row = idx[b * TN + p];
        const float* xr = x + ((size_t)(b * TN + row)) * DIM;
#pragma unroll
        for (int i = 0; i < 24; ++i) acc[i] += xr[lane + 32 * i];
    }
    float s1 = 0.f, s2 = 0.f;
#pragma unroll
    for (int i = 0; i < 24; ++i) {
        float t = acc[i] * invw; acc[i] = t;
        s1 += t; s2 += t * t;
    }
#pragma unroll
    for (int off = 16; off > 0; off >>= 1) {
        s1 += __shfl_xor(s1, off, 32);
        s2 += __shfl_xor(s2, off, 32);
    }
    const float mu  = s1 * (1.0f / DIM);
    const float var = s2 * (1.0f / DIM) - mu * mu;
    const float rstd = rsqrtf(var + EPS);
    float dot = 0.f;
#pragma unroll
    for (int i = 0; i < 24; ++i) {
        int d = lane + 32 * i;
        float y = (acc[i] - mu) * rstd * ln_g[d] + ln_b[d];
        dot += y * w[(size_t)a * DIM + d];
    }
#pragma unroll
    for (int off = 16; off > 0; off >>= 1) dot += __shfl_xor(dot, off, 32);
    if (lane == 0) logits[b * ATTR + a] = dot + bb[a];
}

// ---------------------------------------------------------------------------
// K5: training-mode BatchNorm1d over the batch dim. One block per attribute.
// ---------------------------------------------------------------------------
__global__ void k5_bn(const float* __restrict__ logits,
                      const float* __restrict__ g,
                      const float* __restrict__ be,
                      float* __restrict__ out) {
    __shared__ float buf[64];
    __shared__ float buf2[64];
    const int a = blockIdx.x;
    const int b = threadIdx.x;
    float v = logits[b * ATTR + a];
    buf[b] = v; buf2[b] = v * v;
    __syncthreads();
    for (int off = 32; off > 0; off >>= 1) {
        if (b < off) { buf[b] += buf[b + off]; buf2[b] += buf2[b + off]; }
        __syncthreads();
    }
    float mean = buf[0] * (1.0f / 64.0f);
    float var  = buf2[0] * (1.0f / 64.0f) - mean * mean;
    out[b * ATTR + a] = (v - mean) * rsqrtf(var + EPS) * g[a] + be[a];
}

// ---------------------------------------------------------------------------
extern "C" void kernel_launch(void* const* d_in, const int* in_sizes, int n_in,
                              void* d_out, int out_size, void* d_ws, size_t ws_size,
                              hipStream_t stream) {
    const float* x     = (const float*)d_in[0];  // [64,6,128,768]
    const float* ln_g  = (const float*)d_in[1];
    const float* ln_b  = (const float*)d_in[2];
    const float* w     = (const float*)d_in[3];  // [51,768]
    const float* battr = (const float*)d_in[4];
    const float* bn_g  = (const float*)d_in[5];
    const float* bn_b  = (const float*)d_in[6];
    float* out = (float*)d_out;

    // workspace layout (all offsets 256B-aligned)
    char* ws = (char*)d_ws;
    const size_t XN_B     = (size_t)BATCH * TN * DIM * sizeof(_Float16); // 72 MB
    const size_t SC_B     = (size_t)BATCH * TN * sizeof(float);
    const size_t KEEP_B   = (size_t)BATCH * TN * sizeof(int);
    const size_t IDX_B    = (size_t)BATCH * TN * sizeof(int);
    _Float16* xn   = (_Float16*)(ws);
    float* scores  = (float*)(ws + XN_B);
    int*   keep    = (int*)(ws + XN_B + SC_B);
    int*   idx     = (int*)(ws + XN_B + SC_B + KEEP_B);
    int*   counts  = (int*)(ws + XN_B + SC_B + KEEP_B + IDX_B);
    int*   Lmax    = (int*)(ws + XN_B + SC_B + KEEP_B + IDX_B + 256);
    float* logits  = (float*)(ws + XN_B + SC_B + KEEP_B + IDX_B + 512);

    hipMemsetAsync(Lmax, 0, sizeof(int), stream);

    // K1: 8 tokens per 256-thread block
    k1_normalize<<<(BATCH * TN) / 8, 256, 0, stream>>>(x, xn, scores);

    // K2: one 16-row tile per block, 8 waves sweep the 48 column tiles
    k2_sim_keep<<<BATCH * NT, 256, 0, stream>>>(xn, keep);

    // K3: one block per sample
    k3_compact<<<BATCH, 256, 0, stream>>>(scores, keep, idx, counts, Lmax);

    // K4: one wave per (sample, attribute)
    k4_head<<<BATCH * ATTR, 32, 0, stream>>>(x, idx, counts, Lmax,
                                             ln_g, ln_b, w, battr, logits);

    // K5: one block per attribute
    k5_bn<<<ATTR, 64, 0, stream>>>(logits, bn_g, bn_b, out);
}
